// Simple_RNN_84155589198290
// MI455X (gfx1250) — matrix operand order repaired and checked by reading, not compiled, not run on hardware
//
#include <hip/hip_runtime.h>
#include <cstdint>
#include <cstddef>

// Problem constants (from reference): B,T,D,L,H
#define B_ 256
#define T_ 1000
#define D_ 128
#define L_ 128
#define H_ 64
// yc width = L+D = 256 (GEMM1 K), GEMM1 N = 64, GEMM2 K = 64, GEMM2 N = 128

typedef __attribute__((ext_vector_type(16))) __bf16 v16bf;
typedef __attribute__((ext_vector_type(8)))  float  v8f;

union Frag {
    v16bf v;
    uint4 u[2];
};

// ---------------------------------------------------------------------------
// WMMA wrapper: D = A(16x32 bf16) * B(32x16 bf16) + C(16x16 f32)
// ---------------------------------------------------------------------------
__device__ __forceinline__ v8f bf16mma(v16bf a, v16bf b, v8f c) {
    return __builtin_amdgcn_wmma_f32_16x16x32_bf16(
        /*neg_a=*/false, a, /*neg_b=*/false, b,
        /*c_mod=*/(short)0, c, /*reuse_a=*/false, /*reuse_b=*/false);
}

// A-fragment (16x32 bf16) from a row-major bf16 matrix in LDS.
// ISA layout: lanes 0-15 row=lane, K chunks [k0..k0+7] and [k0+16..k0+23]
//             lanes 16-31 row=lane-16, chunks at +8 (khalf8 handles this).
__device__ __forceinline__ v16bf ldsAfrag(const __bf16* base, int stride, int row, int k0) {
    Frag f;
    f.u[0] = *(const uint4*)(base + row * stride + k0);
    f.u[1] = *(const uint4*)(base + row * stride + k0 + 16);
    return f.v;
}

// B-fragment from pre-packed fragment-major storage: frag*512 + lane*16 bf16.
__device__ __forceinline__ v16bf ldsBfrag(const __bf16* base, int frag, int lane) {
    Frag f;
    const uint4* p = (const uint4*)(base + frag * 512 + lane * 16);
    f.u[0] = p[0];
    f.u[1] = p[1];
    return f.v;
}

// ---------------------------------------------------------------------------
// Pack fp32 weights into bf16 WMMA B-fragment layout (one-time per launch).
// B 32x16 bf16 lane map: lane l holds column n=(l&15), K = (l>=16?8:0)+(j&7)+(j>=8?16:0),
// j=0..15 packed contiguously (VGPR g holds K pair 2g,2g+1).
// W1: [256,64] -> frag(g,kt<8,nt<4).  W2: [64,128] -> frag(g,kt<2,nt<8).
// ---------------------------------------------------------------------------
__global__ void pack_weights_k(const float* __restrict__ uw1, const float* __restrict__ rw1,
                               const float* __restrict__ nw1, const float* __restrict__ uw2,
                               const float* __restrict__ rw2, const float* __restrict__ nw2,
                               __bf16* __restrict__ w1p, __bf16* __restrict__ w2p) {
    const int idx = blockIdx.x * 256 + threadIdx.x;
    const float* w1s[3] = {uw1, rw1, nw1};
    const float* w2s[3] = {uw2, rw2, nw2};

    if (idx < 3 * 8 * 4 * 512) {                 // GEMM1 weights
        const int j    = idx & 15;
        const int l    = (idx >> 4) & 31;
        const int frag = idx >> 9;               // (g*8+kt)*4+nt
        const int nt   = frag & 3;
        const int kt   = (frag >> 2) & 7;
        const int g    = frag >> 5;
        const int k    = kt * 32 + ((l >> 4) << 3) + (j & 7) + ((j >> 3) << 4);
        const int n    = nt * 16 + (l & 15);
        w1p[idx] = (__bf16)w1s[g][k * 64 + n];
    }
    const int idx2 = idx - 3 * 8 * 4 * 512;
    if (idx2 >= 0 && idx2 < 3 * 2 * 8 * 512) {   // GEMM2 weights
        const int j    = idx2 & 15;
        const int l    = (idx2 >> 4) & 31;
        const int frag = idx2 >> 9;              // (g*2+kt)*8+nt
        const int nt   = frag & 7;
        const int kt   = (frag >> 3) & 1;
        const int g    = frag >> 4;
        const int k    = kt * 32 + ((l >> 4) << 3) + (j & 7) + ((j >> 3) << 4);
        const int n    = nt * 16 + (l & 15);
        w2p[idx2] = (__bf16)w2s[g][k * 128 + n];
    }
}

// ---------------------------------------------------------------------------
// time_intervals = time_steps[:,1:] - time_steps[:,:-1]
// ---------------------------------------------------------------------------
__global__ void time_intervals_k(const float* __restrict__ ts, float* __restrict__ out) {
    const int i = blockIdx.x * 256 + threadIdx.x;
    if (i < B_ * (T_ - 1)) {
        const int b = i / (T_ - 1);
        const int t = i - b * (T_ - 1);
        out[i] = ts[b * T_ + t + 1] - ts[b * T_ + t];
    }
}

// ---------------------------------------------------------------------------
// Persistent recurrent kernel. 16 blocks x 128 threads (4 wave32).
// Block bm owns batch rows [bm*16, bm*16+16); state y lives in LDS for all T.
// Wave w owns GEMM1 N-tile w (N=64 -> 4 tiles) and GEMM2 N-tiles {2w,2w+1}.
// ---------------------------------------------------------------------------
__global__ void __launch_bounds__(128, 1)
rnn_persist_k(const float* __restrict__ data,
              const float* __restrict__ ub1, const float* __restrict__ ub2,
              const float* __restrict__ rb1, const float* __restrict__ rb2,
              const float* __restrict__ nb1, const float* __restrict__ nb2,
              const __bf16* __restrict__ w1p, const __bf16* __restrict__ w2p,
              float* __restrict__ out_ylast, float* __restrict__ out_ys) {
    __shared__ __bf16 w1lds[3 * 8 * 4 * 512];  // 96 KB: GEMM1 weight fragments
    __shared__ __bf16 aBuf[16 * 256];          // 8 KB: A-tile [y|xf] (later [y*r|xf])
    __shared__ __bf16 hBuf[3][16 * 64];        // 6 KB: tanh(hidden) per gate
    __shared__ float  yLDS[16 * 128];          // 8 KB: recurrent state (f32)
    __shared__ float  mrow[16];                // mask row-sums

    const int tid    = threadIdx.x;
    const int lane   = tid & 31;
    const int wave   = tid >> 5;
    const int col    = lane & 15;              // N-in-tile / A-row for this lane
    const int khalf8 = (lane >> 4) << 3;       // A chunk base: 0 or 8
    const int m0     = blockIdx.x * 16;        // first batch row of this block

    // Stage GEMM1 weight fragments into LDS (one-time).
    {
        const uint4* src = (const uint4*)w1p;
        uint4*       dst = (uint4*)w1lds;
        for (int i = tid; i < (3 * 8 * 4 * 512) / 8; i += 128) dst[i] = src[i];
    }
    // Zero recurrent state.
    for (int i = tid; i < 16 * 128; i += 128) yLDS[i] = 0.0f;

    // GEMM2 weight fragments: register-resident for the whole T loop.
    Frag w2f[3][2][2];  // [gate][ntLocal][kt]
#pragma unroll
    for (int g = 0; g < 3; ++g)
#pragma unroll
        for (int nl = 0; nl < 2; ++nl)
#pragma unroll
            for (int kt = 0; kt < 2; ++kt) {
                const int frag = (g * 2 + kt) * 8 + (wave * 2 + nl);
                const uint4* p = (const uint4*)(w2p + frag * 512 + lane * 16);
                w2f[g][nl][kt].u[0] = p[0];
                w2f[g][nl][kt].u[1] = p[1];
            }

    // Biases (depend only on this lane's output column).
    const float* b1[3] = {ub1, rb1, nb1};
    const float* b2[3] = {ub2, rb2, nb2};
    float bias1[3], bias2[3][2];
#pragma unroll
    for (int g = 0; g < 3; ++g) {
        bias1[g]    = b1[g][wave * 16 + col];
        bias2[g][0] = b2[g][(wave * 2 + 0) * 16 + col];
        bias2[g][1] = b2[g][(wave * 2 + 1) * 16 + col];
    }

    const int lr = tid >> 3;         // row this thread loads (0..15)
    const int lc = (tid & 7) * 32;   // 32-column group (feat if <128 else mask)

    __syncthreads();

    for (int t = 0; t < T_; ++t) {
        if (tid < 16) mrow[tid] = 0.0f;
        __syncthreads();

        // ---- Stage x_t: features -> bf16 A-tile cols 128..255; mask -> row sums.
        {
            const float* drow = data + ((size_t)(m0 + lr) * T_ + t) * 256 + lc;
            if (t + 1 < T_)
                __builtin_prefetch(drow + 256, 0, 1);   // global_prefetch_b8 next step
            if (lc < 128) {
#pragma unroll
                for (int q = 0; q < 8; ++q) {
                    const float4 v = *(const float4*)(drow + q * 4);
                    __bf16* dst = &aBuf[lr * 256 + 128 + lc + q * 4];
                    dst[0] = (__bf16)v.x; dst[1] = (__bf16)v.y;
                    dst[2] = (__bf16)v.z; dst[3] = (__bf16)v.w;
                }
            } else {
                float msum = 0.0f;
#pragma unroll
                for (int q = 0; q < 8; ++q) {
                    const float4 v = *(const float4*)(drow + q * 4);
                    msum += v.x + v.y + v.z + v.w;
                }
                atomicAdd(&mrow[lr], msum);
            }
        }
        // ---- y (f32) -> bf16 A-tile cols 0..127.
        for (int i = tid; i < 16 * 128; i += 128)
            aBuf[(i >> 7) * 256 + (i & 127)] = (__bf16)yLDS[i];
        __syncthreads();

        // ---- GEMM1 for u,r gates: [16,256] x [256,64], split accum chain in two.
#pragma unroll
        for (int g = 0; g < 2; ++g) {
            v8f acc0, acc1 = {0, 0, 0, 0, 0, 0, 0, 0};
#pragma unroll
            for (int i = 0; i < 8; ++i) acc0[i] = bias1[g];
#pragma unroll
            for (int kt = 0; kt < 8; kt += 2) {
                acc0 = bf16mma(ldsAfrag(aBuf, 256, col, kt * 32 + khalf8),
                               ldsBfrag(w1lds, (g * 8 + kt) * 4 + wave, lane), acc0);
                acc1 = bf16mma(ldsAfrag(aBuf, 256, col, (kt + 1) * 32 + khalf8),
                               ldsBfrag(w1lds, (g * 8 + kt + 1) * 4 + wave, lane), acc1);
            }
#pragma unroll
            for (int i = 0; i < 8; ++i) {
                const int M = i + ((lane >> 4) << 3);
                hBuf[g][M * 64 + wave * 16 + col] = (__bf16)tanhf(acc0[i] + acc1[i]);
            }
        }
        __syncthreads();

        // ---- GEMM2 u: sigmoid, keep in registers until combine.
        v8f uacc[2];
#pragma unroll
        for (int nl = 0; nl < 2; ++nl) {
#pragma unroll
            for (int i = 0; i < 8; ++i) uacc[nl][i] = bias2[0][nl];
#pragma unroll
            for (int kt = 0; kt < 2; ++kt)
                uacc[nl] = bf16mma(ldsAfrag(hBuf[0], 64, col, kt * 32 + khalf8),
                                   w2f[0][nl][kt].v, uacc[nl]);
#pragma unroll
            for (int i = 0; i < 8; ++i)
                uacc[nl][i] = 1.0f / (1.0f + __expf(-uacc[nl][i]));
        }
        // ---- GEMM2 r: sigmoid fused into rebuilding A-tile y-part as y*r.
#pragma unroll
        for (int nl = 0; nl < 2; ++nl) {
            v8f racc;
#pragma unroll
            for (int i = 0; i < 8; ++i) racc[i] = bias2[1][nl];
#pragma unroll
            for (int kt = 0; kt < 2; ++kt)
                racc = bf16mma(ldsAfrag(hBuf[1], 64, col, kt * 32 + khalf8),
                               w2f[1][nl][kt].v, racc);
#pragma unroll
            for (int i = 0; i < 8; ++i) {
                const float r = 1.0f / (1.0f + __expf(-racc[i]));
                const int M = i + ((lane >> 4) << 3);
                const int n = (wave * 2 + nl) * 16 + col;
                aBuf[M * 256 + n] = (__bf16)(yLDS[M * 128 + n] * r);
            }
        }
        __syncthreads();

        // ---- GEMM1 n gate on [y*r | xf].
        {
            v8f acc0, acc1 = {0, 0, 0, 0, 0, 0, 0, 0};
#pragma unroll
            for (int i = 0; i < 8; ++i) acc0[i] = bias1[2];
#pragma unroll
            for (int kt = 0; kt < 8; kt += 2) {
                acc0 = bf16mma(ldsAfrag(aBuf, 256, col, kt * 32 + khalf8),
                               ldsBfrag(w1lds, (2 * 8 + kt) * 4 + wave, lane), acc0);
                acc1 = bf16mma(ldsAfrag(aBuf, 256, col, (kt + 1) * 32 + khalf8),
                               ldsBfrag(w1lds, (2 * 8 + kt + 1) * 4 + wave, lane), acc1);
            }
#pragma unroll
            for (int i = 0; i < 8; ++i) {
                const int M = i + ((lane >> 4) << 3);
                hBuf[2][M * 64 + wave * 16 + col] = (__bf16)tanhf(acc0[i] + acc1[i]);
            }
        }
        __syncthreads();

        // ---- GEMM2 n + gated combine + mask + state update + output store.
#pragma unroll
        for (int nl = 0; nl < 2; ++nl) {
            v8f nacc;
#pragma unroll
            for (int i = 0; i < 8; ++i) nacc[i] = bias2[2][nl];
#pragma unroll
            for (int kt = 0; kt < 2; ++kt)
                nacc = bf16mma(ldsAfrag(hBuf[2], 64, col, kt * 32 + khalf8),
                               w2f[2][nl][kt].v, nacc);
#pragma unroll
            for (int i = 0; i < 8; ++i) {
                const int M = i + ((lane >> 4) << 3);
                const int n = (wave * 2 + nl) * 16 + col;
                const float yprev = yLDS[M * 128 + n];
                const float u = uacc[nl][i];
                float ny = (1.0f - u) * nacc[i] + u * yprev;
                const float m = (mrow[M] > 0.0f) ? 1.0f : 0.0f;
                ny = m * ny + (1.0f - m) * yprev;
                yLDS[M * 128 + n] = ny;
                if (t < T_ - 1)
                    out_ys[((size_t)(m0 + M) * (T_ - 1) + t) * 128 + n] = ny;
            }
        }
        __syncthreads();
    }

    // y_last = final state.
    for (int i = tid; i < 16 * 128; i += 128)
        out_ylast[(size_t)(m0 + (i >> 7)) * 128 + (i & 127)] = yLDS[i];
}

// ---------------------------------------------------------------------------
extern "C" void kernel_launch(void* const* d_in, const int* in_sizes, int n_in,
                              void* d_out, int out_size, void* d_ws, size_t ws_size,
                              hipStream_t stream) {
    (void)in_sizes; (void)n_in; (void)out_size; (void)ws_size;

    const float* data = (const float*)d_in[0];
    const float* ts   = (const float*)d_in[1];
    const float* uw1  = (const float*)d_in[2];
    const float* ub1  = (const float*)d_in[3];
    const float* uw2  = (const float*)d_in[4];
    const float* ub2  = (const float*)d_in[5];
    const float* rw1  = (const float*)d_in[6];
    const float* rb1  = (const float*)d_in[7];
    const float* rw2  = (const float*)d_in[8];
    const float* rb2  = (const float*)d_in[9];
    const float* nw1  = (const float*)d_in[10];
    const float* nb1  = (const float*)d_in[11];
    const float* nw2  = (const float*)d_in[12];
    const float* nb2  = (const float*)d_in[13];

    // Workspace: packed bf16 weight fragments (144 KB total).
    __bf16* w1p = (__bf16*)d_ws;               // 3*8*4*512 elements
    __bf16* w2p = w1p + 3 * 8 * 4 * 512;       // 3*2*8*512 elements

    // Output partitions (flat, return order): y_last | latent_ys[:, :-1] | dt
    float* out_ylast = (float*)d_out;
    float* out_ys    = out_ylast + (size_t)B_ * L_;
    float* out_ti    = out_ys + (size_t)B_ * (T_ - 1) * L_;

    pack_weights_k<<<288, 256, 0, stream>>>(uw1, rw1, nw1, uw2, rw2, nw2, w1p, w2p);
    time_intervals_k<<<(B_ * (T_ - 1) + 255) / 256, 256, 0, stream>>>(ts, out_ti);
    rnn_persist_k<<<16, 128, 0, stream>>>(data, ub1, ub2, rb1, rb2, nb1, nb2,
                                          w1p, w2p, out_ylast, out_ys);
}
